// MinGRUWrapper_7842610283420
// MI455X (gfx1250) — matrix operand order repaired and checked by reading, not compiled, not run on hardware
//
#include <hip/hip_runtime.h>
#include <hip/hip_bf16.h>
#include <math.h>

// ---------------------------------------------------------------------------
// Types for CDNA5 WMMA / TDM
// ---------------------------------------------------------------------------
typedef __attribute__((ext_vector_type(16))) __bf16 v16bf;
typedef __attribute__((ext_vector_type(8)))  float  v8f;
typedef __attribute__((ext_vector_type(4)))  unsigned int u32x4;
typedef __attribute__((ext_vector_type(8)))  int          i32x8;
typedef __attribute__((ext_vector_type(4)))  int          i32x4;

union ABu { v16bf v; uint4 q[2]; };

#define DIM     512
#define DFF     2048
#define SEQ     4096
#define BATCH   8
#define MTOT    (BATCH * SEQ)     // 32768 tokens
#define NCHUNK  32
#define CHLEN   (SEQ / NCHUNK)    // 128
#define BD      (BATCH * DIM)     // 4096 channels

// ---------------------------------------------------------------------------
// Math helpers
// ---------------------------------------------------------------------------
__device__ __forceinline__ unsigned short f2bf(float f) {
    union { float f; unsigned u; } v; v.f = f;
    unsigned u = v.u;
    unsigned r = u + 0x7FFFu + ((u >> 16) & 1u);   // round-to-nearest-even
    return (unsigned short)(r >> 16);
}

__device__ __forceinline__ float softplusf(float x) {
    if (x > 20.f)  return x;
    if (x < -20.f) return expf(x);
    return log1pf(expf(x));
}

__device__ __forceinline__ float log_g_dev(float x) {
    return (x >= 0.f) ? logf(x + 0.5f) : -softplusf(-x);
}

__device__ __forceinline__ float gelu_exact(float x) {
    return 0.5f * x * (1.f + erff(x * 0.70710678118654752f));
}

__device__ __forceinline__ float logaddexpf(float a, float b) {
    float mx = fmaxf(a, b);
    float mn = fminf(a, b);
    if (mx == -__builtin_inff()) return -__builtin_inff();
    return mx + log1pf(expf(mn - mx));
}

// ---------------------------------------------------------------------------
// TDM: async DMA of a 128-row x 32-col bf16 tile (row stride K elems) -> LDS.
// D# per CDNA5 ISA ch.8: group0 = flags/lds_addr/global_addr/type,
// group1 = data_size, tensor dims, tile dims, strides; groups 2/3 = 0 (2-D).
// This toolchain exposes the 6-arg builtin (extra int32x8 group).
// ---------------------------------------------------------------------------
__device__ __forceinline__ void tdm_load_tile(unsigned long long gaddr,
                                              unsigned ldsaddr, int K) {
    u32x4 g0;
    g0[0] = 1u;                                            // count=1, no gather
    g0[1] = ldsaddr;                                       // LDS byte address
    g0[2] = (unsigned)(gaddr & 0xFFFFFFFFull);             // global_addr[31:0]
    g0[3] = (unsigned)((gaddr >> 32) & 0x1FFFFFFull)       // global_addr[56:32]
          | (2u << 30);                                    // type = 2 ("image")
    i32x8 g1;
    g1[0] = 0x00010000;                                    // data_size=1 (2B)
    g1[1] = (int)(((unsigned)K & 0xFFFFu) << 16);          // tensor_dim0 lo
    g1[2] = (int)((((unsigned)K >> 16) & 0xFFFFu)
          | 0x80000000u);                                  // dim0 hi | dim1 lo (32768)
    g1[3] = (int)(32u << 16);                              // tile_dim0 = 32
    g1[4] = 128;                                           // tile_dim1 = 128
    g1[5] = K;                                             // tensor_dim0_stride
    g1[6] = 0;
    g1[7] = 0;
    i32x4 z4 = {};
    i32x8 z8 = {};
    __builtin_amdgcn_tensor_load_to_lds(g0, g1, z4, z4, z8, 0);
}

// ---------------------------------------------------------------------------
// RMSNorm: 8 waves/block, one wave per token (D=512 -> 16 elems/lane, wave32)
// ---------------------------------------------------------------------------
__global__ __launch_bounds__(256)
void rmsnorm_bf16_kernel(const float* __restrict__ x, const float* __restrict__ w,
                         unsigned short* __restrict__ out, int D) {
    const int token = blockIdx.x * 8 + (threadIdx.x >> 5);
    const int lane  = threadIdx.x & 31;
    const float* xr = x + (size_t)token * D;
    float ss = 0.f;
    for (int i = lane; i < D; i += 32) { float v = xr[i]; ss += v * v; }
    for (int off = 16; off > 0; off >>= 1) ss += __shfl_xor(ss, off, 32);
    const float scale = rsqrtf(ss / (float)D + 1e-6f);
    unsigned short* orow = out + (size_t)token * D;
    for (int i = lane; i < D; i += 32) orow[i] = f2bf(xr[i] * scale * w[i]);
}

__global__ __launch_bounds__(256)
void rmsnorm_f32_kernel(const float* __restrict__ x, const float* __restrict__ w,
                        float* __restrict__ out, int D) {
    const int token = blockIdx.x * 8 + (threadIdx.x >> 5);
    const int lane  = threadIdx.x & 31;
    const float* xr = x + (size_t)token * D;
    float ss = 0.f;
    for (int i = lane; i < D; i += 32) { float v = xr[i]; ss += v * v; }
    for (int off = 16; off > 0; off >>= 1) ss += __shfl_xor(ss, off, 32);
    const float scale = rsqrtf(ss / (float)D + 1e-6f);
    float* orow = out + (size_t)token * D;
    for (int i = lane; i < D; i += 32) orow[i] = xr[i] * scale * w[i];
}

// ---------------------------------------------------------------------------
// Pack fp32 weight [K,N] into bf16 WMMA B-fragment layout:
// tile (kt,nt) of 32x16; lane = (n&15) + 16*(k_half); 16 contiguous bf16 per
// lane holding column n, rows k = kt*32 + k_half*16 + j  (j = 0..15).
// ---------------------------------------------------------------------------
__global__ __launch_bounds__(256)
void pack_b_kernel(const float* __restrict__ W, unsigned short* __restrict__ P,
                   int K, int N) {
    const size_t t = (size_t)blockIdx.x * blockDim.x + threadIdx.x;  // over K*N
    if (t >= (size_t)K * N) return;
    const int tilesN = N >> 4;
    const size_t tile = t >> 9;           // 512 elems per 32x16 tile
    const int within  = (int)(t & 511);
    const int lane    = within >> 4;
    const int j       = within & 15;
    const int kt = (int)(tile / tilesN);
    const int nt = (int)(tile % tilesN);
    const int n = nt * 16 + (lane & 15);
    const int k = kt * 32 + (lane >> 4) * 16 + j;
    P[t] = f2bf(W[(size_t)k * N + n]);
}

// ---------------------------------------------------------------------------
// bf16 WMMA GEMM: out[M,N] = A[M,K] * B[K,N] with fused epilogues.
// Block = 256 thr (8 waves), block tile = 128(M) x 64(N), wave tile = 16x64.
// A tile (128x32 bf16, 8 KB) is DMA'd into double-buffered LDS by the TDM
// (wave 0 issues; s_wait_tensorcnt + workgroup barrier publish it); WMMA
// operands come from aligned ds_load_b128. B fragments are 32 B/lane
// contiguous global loads of the packed, L2-resident weights.
// EPI 0: minGRU hidden/gate ->  lg / lc / lv0   (N = 2*halfN)
// EPI 1: +bias, exact GELU  ->  bf16 out        (FFN up)
// EPI 2: +bias +residual    ->  f32 out         (FFN down)
// ---------------------------------------------------------------------------
template<int EPI>
__global__ __launch_bounds__(256)
void gemm_wmma_kernel(const unsigned short* __restrict__ A,
                      const unsigned short* __restrict__ Bp,
                      const int M, const int N, const int K,
                      const float* __restrict__ bias,
                      const float* __restrict__ resid,
                      float* __restrict__ outf,
                      unsigned short* __restrict__ outh,
                      float* __restrict__ lg,
                      float* __restrict__ lcb,
                      float* __restrict__ lvb) {
    __shared__ unsigned short tA[2][128 * 32];   // 2 x 8 KB A tiles

    const int lane = threadIdx.x & 31;
    const int wave = threadIdx.x >> 5;
    const int m0 = blockIdx.x * 128 + wave * 16;
    const int n0 = blockIdx.y * 64;
    const int tilesN = N >> 4;
    const int ntile0 = n0 >> 4;

    const unsigned long long gA =
        (unsigned long long)(size_t)A + ((size_t)blockIdx.x * 128) * (size_t)K * 2ull;
    const unsigned ldsA0 = (unsigned)(size_t)(&tA[0][0]);
    const unsigned ldsA1 = (unsigned)(size_t)(&tA[1][0]);

    if (wave == 0) tdm_load_tile(gA, ldsA0, K);          // prologue: tile 0

    const int lrow = wave * 16 + (lane & 15);            // LDS-local A row
    const int akb  = (lane >> 4) * 16;                   // K sub-base (elems)

    v8f acc0 = {}; v8f acc1 = {}; v8f acc2 = {}; v8f acc3 = {};

    for (int kt = 0; kt < K; kt += 32) {
        if (wave == 0) __builtin_amdgcn_s_wait_tensorcnt(0);
        __syncthreads();                                  // publish LDS tile
        const int cur = (kt >> 5) & 1;
        if (wave == 0 && kt + 32 < K)                     // overlap next DMA
            tdm_load_tile(gA + (size_t)(kt + 32) * 2ull, cur ? ldsA0 : ldsA1, K);

        // --- A fragment from LDS: two aligned 16B ds loads per lane ---
        ABu a;
        const unsigned short* ap = &tA[cur][lrow * 32 + akb];
        a.q[0] = *(const uint4*)(ap);
        a.q[1] = *(const uint4*)(ap + 8);

        // --- B fragments: 32B contiguous per lane from packed layout ---
        const int ktile = kt >> 5;
        const size_t base = (((size_t)ktile * tilesN + ntile0) * 32 + lane) * 16;
        ABu b0, b1, b2, b3;
        b0.q[0] = *(const uint4*)(Bp + base);          b0.q[1] = *(const uint4*)(Bp + base + 8);
        b1.q[0] = *(const uint4*)(Bp + base + 512);    b1.q[1] = *(const uint4*)(Bp + base + 520);
        b2.q[0] = *(const uint4*)(Bp + base + 1024);   b2.q[1] = *(const uint4*)(Bp + base + 1032);
        b3.q[0] = *(const uint4*)(Bp + base + 1536);   b3.q[1] = *(const uint4*)(Bp + base + 1544);

        acc0 = __builtin_amdgcn_wmma_f32_16x16x32_bf16(false, a.v, false, b0.v, (short)0, acc0, false, false);
        acc1 = __builtin_amdgcn_wmma_f32_16x16x32_bf16(false, a.v, false, b1.v, (short)0, acc1, false, false);
        acc2 = __builtin_amdgcn_wmma_f32_16x16x32_bf16(false, a.v, false, b2.v, (short)0, acc2, false, false);
        acc3 = __builtin_amdgcn_wmma_f32_16x16x32_bf16(false, a.v, false, b3.v, (short)0, acc3, false, false);
    }

    // --- Epilogue: D layout: VGPR r, lane<16 -> M=r ; lane>=16 -> M=r+8 ---
    const int mbase = m0 + (lane >> 4) * 8;
    const int nbase = n0 + (lane & 15);
    const int halfN = N >> 1;

    #pragma unroll
    for (int t = 0; t < 4; ++t) {
        v8f acc = (t == 0) ? acc0 : (t == 1) ? acc1 : (t == 2) ? acc2 : acc3;
        const int n = nbase + t * 16;
        #pragma unroll
        for (int r = 0; r < 8; ++r) {
            const int m = mbase + r;
            float v = acc[r];
            if (EPI == 0) {
                if (n < halfN) {
                    lg[(size_t)m * halfN + n] = log_g_dev(v);          // log g(hidden)
                } else {
                    const int d = n - halfN;
                    lcb[(size_t)m * halfN + d] = -softplusf(v);        // log(1-z)
                    lvb[(size_t)m * halfN + d] = -softplusf(-v);       // log(z)
                }
            } else if (EPI == 1) {
                v += bias[n];
                outh[(size_t)m * N + n] = f2bf(gelu_exact(v));
            } else {
                v += bias[n] + resid[(size_t)m * N + n];
                outf[(size_t)m * N + n] = v;
            }
        }
    }
}

// ---------------------------------------------------------------------------
// Chunked Heinsen scan (log-space recurrence log_h = LSE(lc+log_h, lv)).
// Pass A: per (channel, chunk) local scan -> chunk summary; also writes the
//         combined lv = lv0 + lg back in place so pass C reads 2 streams.
// Pass B: sequential combine of 32 chunk summaries per channel -> carry-in.
// Pass C: redo local scan seeded with carry; out = exp(log_h) + residual.
// ---------------------------------------------------------------------------
__global__ __launch_bounds__(256)
void scan_chunks_kernel(const float* __restrict__ lg, const float* __restrict__ lc,
                        float* __restrict__ lv,
                        float* __restrict__ chA, float* __restrict__ chL) {
    const int t  = blockIdx.x * blockDim.x + threadIdx.x;   // BD * NCHUNK
    const int ch = t % BD;
    const int c  = t / BD;
    const int b  = ch / DIM;
    const int d  = ch % DIM;
    size_t i = ((size_t)b * SEQ + (size_t)c * CHLEN) * DIM + d;
    float Acc = 0.f, L = -__builtin_inff();
    for (int s = 0; s < CHLEN; ++s, i += DIM) {
        const float cc = lc[i];
        const float vv = lv[i] + lg[i];
        lv[i] = vv;                                         // fuse for pass C
        Acc += cc;
        L = logaddexpf(cc + L, vv);
    }
    chA[t] = Acc; chL[t] = L;
}

__global__ __launch_bounds__(256)
void scan_carry_kernel(const float* __restrict__ chA, const float* __restrict__ chL,
                       float* __restrict__ carry) {
    const int ch = blockIdx.x * blockDim.x + threadIdx.x;
    if (ch >= BD) return;
    float prev = -__builtin_inff();
    for (int c = 0; c < NCHUNK; ++c) {
        carry[c * BD + ch] = prev;
        prev = logaddexpf(chA[c * BD + ch] + prev, chL[c * BD + ch]);
    }
}

__global__ __launch_bounds__(256)
void scan_final_kernel(const float* __restrict__ lc, const float* __restrict__ lv,
                       const float* __restrict__ carry,
                       const float* __restrict__ xin, float* __restrict__ xout) {
    const int t  = blockIdx.x * blockDim.x + threadIdx.x;
    const int ch = t % BD;
    const int c  = t / BD;
    const int b  = ch / DIM;
    const int d  = ch % DIM;
    size_t i = ((size_t)b * SEQ + (size_t)c * CHLEN) * DIM + d;
    float L = carry[t];
    for (int s = 0; s < CHLEN; ++s, i += DIM) {
        L = logaddexpf(lc[i] + L, lv[i]);
        xout[i] = expf(L) + xin[i];
    }
}

// ---------------------------------------------------------------------------
// Host orchestration
// ---------------------------------------------------------------------------
extern "C" void kernel_launch(void* const* d_in, const int* in_sizes, int n_in,
                              void* d_out, int out_size, void* d_ws, size_t ws_size,
                              hipStream_t stream) {
    (void)in_sizes; (void)n_in; (void)out_size; (void)ws_size;
    const float* x_in       = (const float*)d_in[0];
    const float* whg        = (const float*)d_in[1];
    const float* norm_w     = (const float*)d_in[2];
    const float* ff_norm_w  = (const float*)d_in[3];
    const float* ff_w1      = (const float*)d_in[4];
    const float* ff_b1      = (const float*)d_in[5];
    const float* ff_w2      = (const float*)d_in[6];
    const float* ff_b2      = (const float*)d_in[7];
    const float* final_w    = (const float*)d_in[8];

    char* ws = (char*)d_ws;
    const size_t MD4 = (size_t)MTOT * DIM * 4;       // 64 MiB
    const size_t MD2 = (size_t)MTOT * DIM * 2;       // 32 MiB
    size_t off = 0;
    auto carve = [&](size_t bytes) { char* p = ws + off; off += (bytes + 255) & ~(size_t)255; return p; };

    float*          x_cur = (float*)carve(MD4);
    float*          x_nxt = (float*)carve(MD4);
    unsigned short* xn    = (unsigned short*)carve(MD2);
    float*          lg    = (float*)carve(MD4);
    float*          lcb   = (float*)carve(MD4);
    float*          lvb   = (float*)carve(MD4);
    unsigned short* hff   = (unsigned short*)carve((size_t)MTOT * DFF * 2);  // 128 MiB
    unsigned short* pack  = (unsigned short*)carve((size_t)DFF * DIM * 2);   // 2 MiB (max)
    float*          chA   = (float*)carve((size_t)BD * NCHUNK * 4);
    float*          chL   = (float*)carve((size_t)BD * NCHUNK * 4);
    float*          carry = (float*)carve((size_t)BD * NCHUNK * 4);

    (void)hipMemcpyAsync(x_cur, x_in, MD4, hipMemcpyDeviceToDevice, stream);

    const dim3 blk(256);
    const dim3 gNorm(MTOT / 8);
    const dim3 gScan((BD * NCHUNK) / 256);
    const dim3 gCarry(BD / 256);

    for (int l = 0; l < 4; ++l) {
        // ---- minGRU branch ----
        rmsnorm_bf16_kernel<<<gNorm, blk, 0, stream>>>(x_cur, norm_w + l * DIM, xn, DIM);
        pack_b_kernel<<<(DIM * 2 * DIM) / 256, blk, 0, stream>>>(whg + (size_t)l * DIM * 2 * DIM, pack, DIM, 2 * DIM);
        gemm_wmma_kernel<0><<<dim3(MTOT / 128, (2 * DIM) / 64), blk, 0, stream>>>(
            xn, pack, MTOT, 2 * DIM, DIM, nullptr, nullptr, nullptr, nullptr, lg, lcb, lvb);
        scan_chunks_kernel<<<gScan, blk, 0, stream>>>(lg, lcb, lvb, chA, chL);
        scan_carry_kernel<<<gCarry, blk, 0, stream>>>(chA, chL, carry);
        scan_final_kernel<<<gScan, blk, 0, stream>>>(lcb, lvb, carry, x_cur, x_nxt);

        // ---- FFN branch ----
        rmsnorm_bf16_kernel<<<gNorm, blk, 0, stream>>>(x_nxt, ff_norm_w + l * DIM, xn, DIM);
        pack_b_kernel<<<(DIM * DFF) / 256, blk, 0, stream>>>(ff_w1 + (size_t)l * DIM * DFF, pack, DIM, DFF);
        gemm_wmma_kernel<1><<<dim3(MTOT / 128, DFF / 64), blk, 0, stream>>>(
            xn, pack, MTOT, DFF, DIM, ff_b1 + (size_t)l * DFF, nullptr, nullptr, hff, nullptr, nullptr, nullptr);
        pack_b_kernel<<<(DFF * DIM) / 256, blk, 0, stream>>>(ff_w2 + (size_t)l * DFF * DIM, pack, DFF, DIM);
        gemm_wmma_kernel<2><<<dim3(MTOT / 128, DIM / 64), blk, 0, stream>>>(
            hff, pack, MTOT, DIM, DFF, ff_b2 + (size_t)l * DIM, x_nxt, x_cur, nullptr, nullptr, nullptr, nullptr);
    }

    rmsnorm_f32_kernel<<<gNorm, blk, 0, stream>>>(x_cur, final_w, (float*)d_out, DIM);
}
